// MultiHeadAttention_70901320122832
// MI455X (gfx1250) — matrix operand order repaired and checked by reading, not compiled, not run on hardware
//
#include <hip/hip_runtime.h>
#include <hip/hip_bf16.h>

// ---------------------------------------------------------------------------
// MHA forward for MI455X (gfx1250), bf16 WMMA path + async global->LDS staging.
//   B=2, S=2048, D_MODEL=1024, H=16, DK=64
// Pipeline:
//   1) convert x, Wq, Wk, Wv, Wo fp32 -> bf16 (workspace)
//   2) LDS-staged double-buffered GEMMs (async-to-LDS engine, ASYNCcnt):
//        Q = x@Wq^T+bq -> (b,h,s,d)   K = x@Wk^T+bk -> (b,h,s,d)
//        V = x@Wv^T+bv -> (b,h,d,s)
//   3) flash attention per (b,h): K/V tiles async-staged to LDS once per
//      block (8 waves share), online softmax, WMMA QK^T and PV
//   4) GEMM: out = attn@Wo^T+bo -> fp32 d_out
// ---------------------------------------------------------------------------

typedef __attribute__((ext_vector_type(16))) __bf16 v16bf;
typedef __attribute__((ext_vector_type(8)))  __bf16 v8bf;
typedef __attribute__((ext_vector_type(8)))  float  v8f;

#define D_MODEL 1024
#define NHEADS  16
#define DK      64
#define BATCH   2
#define SEQ     2048

#define LDSROW  40   // GEMM K-slab padded row (80B stride = 20 banks)
#define KROW    72   // attn K-tile padded row (144B stride = 36 banks)
#define VROW    40   // attn V-tile padded row (80B stride = 20 banks)

__device__ __forceinline__ unsigned short f2bf(float f) {
    unsigned int u = __builtin_bit_cast(unsigned int, f);
    u += 0x7FFFu + ((u >> 16) & 1u);   // round to nearest even
    return (unsigned short)(u >> 16);
}

__device__ __forceinline__ v8f wmma_bf16(v16bf a, v16bf b, v8f c) {
    return __builtin_amdgcn_wmma_f32_16x16x32_bf16(
        false, a, false, b, (short)0, c, false, false);
}

// A-operand fragment (16 rows x 32 K), source row-major (rows x ld).
__device__ __forceinline__ v16bf load_a_frag(const unsigned short* base, int ld,
                                             int k0, int lane) {
    int r = lane & 15, hh = lane >> 4;
    const unsigned short* p = base + r * ld + k0 + hh * 8;
    v8bf c0 = *reinterpret_cast<const v8bf*>(p);
    v8bf c1 = *reinterpret_cast<const v8bf*>(p + 16);
    v16bf f;
#pragma unroll
    for (int i = 0; i < 8; ++i) { f[i] = c0[i]; f[i + 8] = c1[i]; }
    return f;
}

// B-operand fragment (32 K x 16 cols), source stored as (N x K) row-major.
__device__ __forceinline__ v16bf load_b_frag(const unsigned short* base, int ld,
                                             int k0, int lane) {
    int r = lane & 15, hh = lane >> 4;
    const unsigned short* p = base + r * ld + k0 + hh * 8;
    v8bf c0 = *reinterpret_cast<const v8bf*>(p);
    v8bf c1 = *reinterpret_cast<const v8bf*>(p + 16);
    v16bf f;
#pragma unroll
    for (int i = 0; i < 8; ++i) { f[i] = c0[i]; f[i + 8] = c1[i]; }
    return f;
}

// ---- gfx1250 async global->LDS copy (tracked by ASYNCcnt) ------------------
__device__ __forceinline__ unsigned lds_byte_off(const void* p) {
    return (unsigned)(unsigned long long)p;   // low 32 bits = LDS address
}

__device__ __forceinline__ void async_ld_b128(unsigned lds_off, const void* gptr) {
    asm volatile("global_load_async_to_lds_b128 %0, %1, off"
                 :: "v"(lds_off), "v"(gptr)
                 : "memory");
}

__device__ __forceinline__ void wait_async_le4() {
    asm volatile("s_wait_asynccnt 0x4" ::: "memory");
}
__device__ __forceinline__ void wait_async_le2() {
    asm volatile("s_wait_asynccnt 0x2" ::: "memory");
}
__device__ __forceinline__ void wait_async_0() {
    asm volatile("s_wait_asynccnt 0x0" ::: "memory");
}

// ---------------------------------------------------------------------------
__global__ void f32_to_bf16_kernel(const float* __restrict__ in,
                                   unsigned short* __restrict__ out, int n) {
    int i = blockIdx.x * blockDim.x + threadIdx.x;
    if (i < n) out[i] = f2bf(in[i]);
}

// ---------------------------------------------------------------------------
// C(MxN) = A(MxK) @ W(NxK)^T + bias.
// Block = 256 threads (8 waves as 4x2); block tile 128(M) x 128(N);
// K stepped by 32 with double-buffered async-to-LDS staging.
__global__ __launch_bounds__(256) void gemm_bf16_kernel(
    const unsigned short* __restrict__ A,
    const unsigned short* __restrict__ W,
    const float* __restrict__ bias,
    float* __restrict__ Cf,
    unsigned short* __restrict__ Cb,
    int M, int N, int K, int mode) {
    __shared__ __attribute__((aligned(16))) unsigned short sA[2][128 * LDSROW];
    __shared__ __attribute__((aligned(16))) unsigned short sB[2][128 * LDSROW];

    int tid  = threadIdx.x;
    int wave = tid >> 5, lane = tid & 31;
    int tiles_n = N >> 7;
    int bm = blockIdx.x / tiles_n;
    int bn = blockIdx.x - bm * tiles_n;
    int row0 = bm << 7, col0 = bn << 7;
    int wm = wave >> 1, wn = wave & 1;

    v8f acc[2][4];
#pragma unroll
    for (int mi = 0; mi < 2; ++mi)
#pragma unroll
        for (int t = 0; t < 4; ++t) acc[mi][t] = v8f{};

    const int c0 = tid * 2;
    const int srow0 = c0 >> 2,       sko0 = (c0 & 3) * 8;
    const int srow1 = (c0 + 1) >> 2, sko1 = ((c0 + 1) & 3) * 8;

    auto stage = [&](int buf, int k0) {
        async_ld_b128(lds_byte_off(&sA[buf][srow0 * LDSROW + sko0]),
                      A + (size_t)(row0 + srow0) * K + k0 + sko0);
        async_ld_b128(lds_byte_off(&sA[buf][srow1 * LDSROW + sko1]),
                      A + (size_t)(row0 + srow1) * K + k0 + sko1);
        async_ld_b128(lds_byte_off(&sB[buf][srow0 * LDSROW + sko0]),
                      W + (size_t)(col0 + srow0) * K + k0 + sko0);
        async_ld_b128(lds_byte_off(&sB[buf][srow1 * LDSROW + sko1]),
                      W + (size_t)(col0 + srow1) * K + k0 + sko1);
    };

    const int NK = K >> 5;
    stage(0, 0);
    for (int ks = 0; ks < NK; ++ks) {
        int buf = ks & 1;
        if (ks + 1 < NK) {
            stage(buf ^ 1, (ks + 1) << 5);
            wait_async_le4();
        } else {
            wait_async_0();
        }
        __syncthreads();

        const unsigned short* sAb = &sA[buf][(wm * 32) * LDSROW];
        const unsigned short* sBb = &sB[buf][(wn * 64) * LDSROW];
        v16bf af0 = load_a_frag(sAb, LDSROW, 0, lane);
        v16bf af1 = load_a_frag(sAb + 16 * LDSROW, LDSROW, 0, lane);
#pragma unroll
        for (int t = 0; t < 4; ++t) {
            v16bf bf = load_b_frag(sBb + t * 16 * LDSROW, LDSROW, 0, lane);
            acc[0][t] = wmma_bf16(af0, bf, acc[0][t]);
            acc[1][t] = wmma_bf16(af1, bf, acc[1][t]);
        }
        __syncthreads();
    }

    int r = lane & 15, hh = lane >> 4;
#pragma unroll
    for (int mi = 0; mi < 2; ++mi) {
#pragma unroll
        for (int t = 0; t < 4; ++t) {
            int n = col0 + wn * 64 + t * 16 + r;
            float bv = bias ? bias[n] : 0.0f;
#pragma unroll
            for (int i = 0; i < 8; ++i) {
                int m = row0 + wm * 32 + mi * 16 + hh * 8 + i;
                float val = acc[mi][t][i] + bv;
                if (mode == 0) {
                    Cf[(size_t)m * N + n] = val;
                } else {
                    int b = m >> 11, s = m & (SEQ - 1);
                    int h = n >> 6, d = n & (DK - 1);
                    size_t idx;
                    if (mode == 1)
                        idx = (((size_t)(b * NHEADS + h) * SEQ) + s) * DK + d;
                    else
                        idx = (((size_t)(b * NHEADS + h) * DK) + d) * SEQ + s;
                    Cb[idx] = f2bf(val);
                }
            }
        }
    }
}

// ---------------------------------------------------------------------------
// Flash attention. Block = 256 threads = 8 waves; each wave owns 16 query
// rows (block: 128 q-rows) of one (b,h). K/V 32-key tiles are async-staged
// into LDS once per block and shared by all 8 waves; double buffered.
__global__ __launch_bounds__(256) void attn_kernel(
    const unsigned short* __restrict__ Q,    // (B*H, S, DK) bf16
    const unsigned short* __restrict__ Km,   // (B*H, S, DK) bf16
    const unsigned short* __restrict__ Vt,   // (B*H, DK, S) bf16
    unsigned short* __restrict__ O) {        // (B, S, D_MODEL) bf16
    __shared__ __attribute__((aligned(16))) unsigned short sK[2][32 * KROW];
    __shared__ __attribute__((aligned(16))) unsigned short sV[2][64 * VROW];
    __shared__ __attribute__((aligned(16))) unsigned short pbuf[8 * 16 * 32];

    int tid  = threadIdx.x;
    int wave = tid >> 5, lane = tid & 31;
    int qblk = blockIdx.x & ((SEQ / 128) - 1);    // 16 q-blocks per head
    int bh   = blockIdx.x >> 4;                   // 0..B*H-1
    int qrow0 = qblk * 128 + wave * 16;

    const unsigned short* Qb = Q  + (size_t)bh * SEQ * DK + (size_t)qrow0 * DK;
    const unsigned short* Kb = Km + (size_t)bh * SEQ * DK;
    const unsigned short* Vb = Vt + (size_t)bh * DK * SEQ;
    unsigned short* lds = pbuf + wave * (16 * 32);

    v16bf qa0 = load_a_frag(Qb, DK, 0, lane);
    v16bf qa1 = load_a_frag(Qb, DK, 32, lane);

    v8f o0 = {}, o1 = {}, o2 = {}, o3 = {};
    float mrow[8], lrow[8];
#pragma unroll
    for (int i = 0; i < 8; ++i) { mrow[i] = -1e30f; lrow[i] = 0.0f; }
    const float sm_scale = 0.125f;   // 1/sqrt(DK)

    int r = lane & 15, hh = lane >> 4;

    // staging map: K tile 32x64 -> 256 16B chunks; V tile 64x32 -> 256 chunks
    const int krow = tid >> 3, kko = (tid & 7) * 8;    // K: row 0..31, k 0..56
    const int vrow = tid >> 2, vko = (tid & 3) * 8;    // V: row 0..63, k 0..24

    auto stage = [&](int buf, int kb) {
        async_ld_b128(lds_byte_off(&sK[buf][krow * KROW + kko]),
                      Kb + (size_t)(kb + krow) * DK + kko);
        async_ld_b128(lds_byte_off(&sV[buf][vrow * VROW + vko]),
                      Vb + (size_t)vrow * SEQ + kb + vko);
    };

    const int NT = SEQ / 32;                      // 64 key tiles
    stage(0, 0);
    for (int kt = 0; kt < NT; ++kt) {
        int buf = kt & 1;
        int kb = kt * 32;
        if (kt + 1 < NT) {
            stage(buf ^ 1, kb + 32);
            wait_async_le2();
        } else {
            wait_async_0();
        }
        __syncthreads();                          // (a) K/V tile visible

        // ---- scores from LDS K tile ----
        v16bf kf00 = load_b_frag(&sK[buf][0],         KROW, 0,  lane);
        v16bf kf01 = load_b_frag(&sK[buf][0],         KROW, 32, lane);
        v16bf kf10 = load_b_frag(&sK[buf][16 * KROW], KROW, 0,  lane);
        v16bf kf11 = load_b_frag(&sK[buf][16 * KROW], KROW, 32, lane);
        v8f s0 = {}, s1 = {};
        s0 = wmma_bf16(qa0, kf00, s0);
        s0 = wmma_bf16(qa1, kf01, s0);
        s1 = wmma_bf16(qa0, kf10, s1);
        s1 = wmma_bf16(qa1, kf11, s1);

        float p0[8], p1[8], mnew[8];
#pragma unroll
        for (int i = 0; i < 8; ++i) { p0[i] = s0[i] * sm_scale; p1[i] = s1[i] * sm_scale; }

#pragma unroll
        for (int i = 0; i < 8; ++i) {
            float t = fmaxf(p0[i], p1[i]);
            t = fmaxf(t, __shfl_xor(t, 1, 32));
            t = fmaxf(t, __shfl_xor(t, 2, 32));
            t = fmaxf(t, __shfl_xor(t, 4, 32));
            t = fmaxf(t, __shfl_xor(t, 8, 32));
            mnew[i] = fmaxf(mrow[i], t);
        }
#pragma unroll
        for (int i = 0; i < 8; ++i) {
            p0[i] = __expf(p0[i] - mnew[i]);
            p1[i] = __expf(p1[i] - mnew[i]);
            float t = p0[i] + p1[i];
            t += __shfl_xor(t, 1, 32);
            t += __shfl_xor(t, 2, 32);
            t += __shfl_xor(t, 4, 32);
            t += __shfl_xor(t, 8, 32);
            float sc = __expf(mrow[i] - mnew[i]);
            lrow[i] = lrow[i] * sc + t;
            mrow[i] = mnew[i];
            o0[i] *= sc; o1[i] *= sc; o2[i] *= sc; o3[i] *= sc;
        }

        // ---- stage P (C-layout) into per-wave LDS tile, row-major 16x32 ----
#pragma unroll
        for (int i = 0; i < 8; ++i) {
            int row = hh * 8 + i;
            lds[row * 32 + r]      = f2bf(p0[i]);
            lds[row * 32 + 16 + r] = f2bf(p1[i]);
        }
        __syncthreads();                          // (b) P visible / ordered
        v16bf pa = load_a_frag(lds, 32, 0, lane);

        // ---- O += P @ V from LDS V tile ----
        v16bf vb0 = load_b_frag(&sV[buf][ 0 * VROW], VROW, 0, lane);
        v16bf vb1 = load_b_frag(&sV[buf][16 * VROW], VROW, 0, lane);
        v16bf vb2 = load_b_frag(&sV[buf][32 * VROW], VROW, 0, lane);
        v16bf vb3 = load_b_frag(&sV[buf][48 * VROW], VROW, 0, lane);
        o0 = wmma_bf16(pa, vb0, o0);
        o1 = wmma_bf16(pa, vb1, o1);
        o2 = wmma_bf16(pa, vb2, o2);
        o3 = wmma_bf16(pa, vb3, o3);
        __syncthreads();                          // (c) reads done pre-restage
    }

    int b = bh / NHEADS, head = bh % NHEADS;
#pragma unroll
    for (int i = 0; i < 8; ++i) {
        float inv = 1.0f / lrow[i];
        int srow = qrow0 + hh * 8 + i;
        size_t base = ((size_t)b * SEQ + srow) * D_MODEL + head * DK;
        O[base + r +  0] = f2bf(o0[i] * inv);
        O[base + r + 16] = f2bf(o1[i] * inv);
        O[base + r + 32] = f2bf(o2[i] * inv);
        O[base + r + 48] = f2bf(o3[i] * inv);
    }
}

// ---------------------------------------------------------------------------
extern "C" void kernel_launch(void* const* d_in, const int* in_sizes, int n_in,
                              void* d_out, int out_size, void* d_ws, size_t ws_size,
                              hipStream_t stream) {
    const float* x  = (const float*)d_in[0];
    const float* Wq = (const float*)d_in[1];
    const float* bq = (const float*)d_in[2];
    const float* Wk = (const float*)d_in[3];
    const float* bk = (const float*)d_in[4];
    const float* Wv = (const float*)d_in[5];
    const float* bv = (const float*)d_in[6];
    const float* Wo = (const float*)d_in[7];
    const float* bo = (const float*)d_in[8];
    float* out = (float*)d_out;

    char* ws = (char*)d_ws;
    const size_t MB = (size_t)1 << 20;
    unsigned short* xb  = (unsigned short*)(ws + 0 * MB);   // 8 MB
    unsigned short* wqb = (unsigned short*)(ws + 8 * MB);   // 2 MB
    unsigned short* wkb = (unsigned short*)(ws + 10 * MB);  // 2 MB
    unsigned short* wvb = (unsigned short*)(ws + 12 * MB);  // 2 MB
    unsigned short* wob = (unsigned short*)(ws + 14 * MB);  // 2 MB
    unsigned short* qb  = (unsigned short*)(ws + 16 * MB);  // 8 MB (b,h,s,d)
    unsigned short* kbm = (unsigned short*)(ws + 24 * MB);  // 8 MB (b,h,s,d)
    unsigned short* vtb = (unsigned short*)(ws + 32 * MB);  // 8 MB (b,h,d,s)
    unsigned short* ab  = (unsigned short*)(ws + 40 * MB);  // 8 MB (b,s,e)

    const int M = BATCH * SEQ;        // 4096
    const int N = D_MODEL;            // 1024
    const int K = D_MODEL;            // 1024

    int nx = M * D_MODEL;
    int nw = D_MODEL * D_MODEL;
    f32_to_bf16_kernel<<<(nx + 255) / 256, 256, 0, stream>>>(x,  xb,  nx);
    f32_to_bf16_kernel<<<(nw + 255) / 256, 256, 0, stream>>>(Wq, wqb, nw);
    f32_to_bf16_kernel<<<(nw + 255) / 256, 256, 0, stream>>>(Wk, wkb, nw);
    f32_to_bf16_kernel<<<(nw + 255) / 256, 256, 0, stream>>>(Wv, wvb, nw);
    f32_to_bf16_kernel<<<(nw + 255) / 256, 256, 0, stream>>>(Wo, wob, nw);

    int blocks = (M >> 7) * (N >> 7);         // 128x128 block tiles -> 256
    gemm_bf16_kernel<<<blocks, 256, 0, stream>>>(xb, wqb, bq, nullptr, qb,  M, N, K, 1);
    gemm_bf16_kernel<<<blocks, 256, 0, stream>>>(xb, wkb, bk, nullptr, kbm, M, N, K, 1);
    gemm_bf16_kernel<<<blocks, 256, 0, stream>>>(xb, wvb, bv, nullptr, vtb, M, N, K, 2);

    attn_kernel<<<BATCH * NHEADS * (SEQ / 128), 256, 0, stream>>>(qb, kbm, vtb, ab);

    gemm_bf16_kernel<<<blocks, 256, 0, stream>>>(ab, wob, bo, out, nullptr, M, N, K, 0);
}
